// MultiHeadAttention_5102421148449
// MI455X (gfx1250) — matrix operand order repaired and checked by reading, not compiled, not run on hardware
//
#include <hip/hip_runtime.h>
#include <hip/hip_bf16.h>

// MHA forward for B=4,S=2048,D=1024,H=16,DH=64 on gfx1250 (MI455X).
// f16 WMMA for all 4 matmuls; GEMMs and attention use async global->LDS
// double buffering (ASYNCcnt), manually unrolled x2 so buffer selection is
// static and WMMA accumulators stay in place.

#define B_ 4
#define S_ 2048
#define D_ 1024
#define H_ 16
#define DH_ 64

typedef __attribute__((ext_vector_type(16))) _Float16 v16h;
typedef __attribute__((ext_vector_type(8)))  _Float16 v8h;
typedef __attribute__((ext_vector_type(8)))  float    v8f;

union V16U { v16h v; v8h h[2]; };

__device__ __forceinline__ v8f zero8() {
  v8f z;
#pragma unroll
  for (int i = 0; i < 8; ++i) z[i] = 0.0f;
  return z;
}

__device__ __forceinline__ v8f wmma32(v16h a, v16h b, v8f c) {
  return __builtin_amdgcn_wmma_f32_16x16x32_f16(false, a, false, b, (short)0, c,
                                                false, false);
}

// CDNA5 async copy: 16B per lane, global -> LDS, tracked by ASYNCcnt.
__device__ __forceinline__ void async_b128(unsigned int lds_off, const void* gaddr) {
  asm volatile("global_load_async_to_lds_b128 %0, %1, off"
               :
               : "v"(lds_off), "v"((unsigned long long)(uintptr_t)gaddr)
               : "memory");
}

__device__ __forceinline__ void wait_async0() {
  asm volatile("s_wait_asynccnt 0x0" ::: "memory");
}

// A fragment 16x32 f16, matrix row-major [row][k], stride ldm (halves).
__device__ __forceinline__ v16h load_a16x32(const _Float16* base, int ldm,
                                            int row, int k0) {
  int lane = threadIdx.x & 31;
  const _Float16* p =
      base + (size_t)(row + (lane & 15)) * ldm + k0 + ((lane >> 4) << 3);
  V16U u;
  u.h[0] = *(const v8h*)(p);
  u.h[1] = *(const v8h*)(p + 16);
  return u.v;
}

// B fragment 32x16 f16 (K x N), storage is B^T row-major (row = n, contiguous k).
__device__ __forceinline__ v16h load_b32x16(const _Float16* baseT, int ldn,
                                            int col, int k0) {
  int lane = threadIdx.x & 31;
  const _Float16* p =
      baseT + (size_t)(col + (lane & 15)) * ldn + k0 + ((lane >> 4) << 3);
  V16U u;
  u.h[0] = *(const v8h*)(p);
  u.h[1] = *(const v8h*)(p + 16);
  return u.v;
}

// ---------------- conversion kernels ----------------

__global__ void k_cvt_f16(const float* __restrict__ in, _Float16* __restrict__ out,
                          int n) {
  int i = blockIdx.x * 256 + threadIdx.x;
  if (i < n) out[i] = (_Float16)in[i];
}

// in: K x N f32 row-major ; out: N x K f16 row-major (i.e. B^T)
__global__ void k_transpose_f16(const float* __restrict__ in,
                                _Float16* __restrict__ out, int K, int N) {
  int i = blockIdx.x * 256 + threadIdx.x;
  if (i >= K * N) return;
  int nrow = i / K;
  int k = i - nrow * K;
  out[i] = (_Float16)in[(size_t)k * N + nrow];
}

// ---------------- shared GEMM mainloop ----------------
// Block: 256 threads = 8 waves (4 in M x 2 in N). Block tile 64(M) x 128(N).
// K = 1024, k-step 32, double-buffered async staging, wave tile 16x64.

#define TA_ELEM (64 * 40)
#define TB_ELEM (128 * 40)

__device__ __forceinline__ void gemm_step(const _Float16* tac, const _Float16* tbc,
                                          int mw, int nw, v8f acc[4]) {
  v16h a = load_a16x32(tac, 40, mw * 16, 0);
#pragma unroll
  for (int t = 0; t < 4; ++t) {
    v16h b = load_b32x16(tbc, 40, nw * 64 + t * 16, 0);
    acc[t] = wmma32(a, b, acc[t]);
  }
}

__device__ __forceinline__ void gemm_mainloop(const _Float16* __restrict__ Ag,
                                              const _Float16* __restrict__ Bg,
                                              int m0blk, int n0blk,
                                              _Float16* tA, _Float16* tB,
                                              int mw, int nw, v8f acc[4]) {
  int tid = threadIdx.x;

  int arow = tid >> 2, achk = tid & 3;                      // 64 rows x 4 chunks
  const _Float16* aga = Ag + (size_t)(m0blk + arow) * D_ + achk * 8;
  unsigned int al = (unsigned int)(uintptr_t)tA + (unsigned int)(arow * 80 + achk * 16);

  int brow = tid >> 1, bchk = (tid & 1) * 2;                // 128 rows x 4 chunks
  const _Float16* bga = Bg + (size_t)(n0blk + brow) * D_ + bchk * 8;
  unsigned int bl = (unsigned int)(uintptr_t)tB + (unsigned int)(brow * 80 + bchk * 16);

  const unsigned int ABYTES = TA_ELEM * 2;
  const unsigned int BBYTES = TB_ELEM * 2;

  // prologue: stage k-step 0 into buffer 0
  async_b128(al, aga);
  async_b128(bl, bga);
  async_b128(bl + 16, bga + 8);

  for (int it = 0; it < D_ / 32; it += 2) {
    // ---- half A: compute buf0, stage buf1 for it+1 (always valid) ----
    wait_async0();
    __syncthreads();
    {
      int ko = (it + 1) * 32;
      async_b128(al + ABYTES, aga + ko);
      async_b128(bl + BBYTES, bga + ko);
      async_b128(bl + BBYTES + 16, bga + ko + 8);
    }
    gemm_step(tA, tB, mw, nw, acc);

    // ---- half B: compute buf1, stage buf0 for it+2 ----
    wait_async0();
    __syncthreads();
    if (it + 2 < D_ / 32) {
      int ko = (it + 2) * 32;
      async_b128(al, aga + ko);
      async_b128(bl, bga + ko);
      async_b128(bl + 16, bga + ko + 8);
    }
    gemm_step(tA + TA_ELEM, tB + TB_ELEM, mw, nw, acc);
  }
}

// ---------------- QKV projection GEMM ----------------

__global__ void k_gemm_qkv(const _Float16* __restrict__ xh,
                           const _Float16* __restrict__ wT,
                           const float* __restrict__ bias,
                           _Float16* __restrict__ qh, _Float16* __restrict__ kh,
                           _Float16* __restrict__ vt) {
  __shared__ __align__(16) _Float16 tA[2 * TA_ELEM];
  __shared__ __align__(16) _Float16 tB[2 * TB_ELEM];

  int lane = threadIdx.x & 31;
  int wave = threadIdx.x >> 5;
  int mw = wave & 3, nw = wave >> 2;
  int m0blk = blockIdx.y * 64;
  int n0blk = blockIdx.x * 128;

  v8f acc[4];
#pragma unroll
  for (int t = 0; t < 4; ++t) acc[t] = zero8();

  gemm_mainloop(xh, wT, m0blk, n0blk, tA, tB, mw, nw, acc);

  int half = lane >> 4, n16 = lane & 15;
  int m0 = m0blk + mw * 16;
  int n0 = n0blk + nw * 64;
#pragma unroll
  for (int t = 0; t < 4; ++t) {
    int n = n0 + t * 16 + n16;
    int which = n >> 10;        // 0=q 1=k 2=v
    int d = n & (D_ - 1);
    int h = d >> 6;
    int dh = d & (DH_ - 1);
    float bv = bias[n];
#pragma unroll
    for (int r = 0; r < 8; ++r) {
      int m = m0 + r + half * 8;
      int b = m >> 11;
      int s = m & (S_ - 1);
      _Float16 hv = (_Float16)(acc[t][r] + bv);
      size_t bh = (size_t)b * H_ + h;
      if (which == 0)
        qh[(bh * S_ + s) * DH_ + dh] = hv;
      else if (which == 1)
        kh[(bh * S_ + s) * DH_ + dh] = hv;
      else
        vt[(bh * DH_ + dh) * S_ + s] = hv;
    }
  }
}

// ---------------- flash attention ----------------
// grid: (S/64, B*H), block: 128 (4 waves); each wave owns 16 query rows.
// K tile (32 keys x 64 dh, stride 80 halves) and V^T tile (64 dh x 32 keys,
// stride 40 halves) async-staged into double-buffered LDS shared by 4 waves;
// P (16x32) re-laid out through wave-private LDS for the PV WMMA.

#define KT_ELEM (32 * 80)   // 5120 B per buffer
#define VT_ELEM (64 * 40)   // 5120 B per buffer

__device__ __forceinline__ void attn_step(const _Float16* tkc, const _Float16* tvc,
                                          int kb, int q0, int half, int n16,
                                          const int* mp, _Float16* pl,
                                          v16h aql, v16h aqh2, v8f o[4],
                                          float rowm[8], float rowl[8]) {
  // S = Q * K^T (two 16x16 tiles over 32 keys, k-dim = DH = 64)
  v8f s0 = zero8(), s1 = zero8();
  s0 = wmma32(aql, load_b32x16(tkc, 80, 0, 0), s0);
  s0 = wmma32(aqh2, load_b32x16(tkc, 80, 0, 32), s0);
  s1 = wmma32(aql, load_b32x16(tkc, 80, 16, 0), s1);
  s1 = wmma32(aqh2, load_b32x16(tkc, 80, 16, 32), s1);

  float sv0[8], sv1[8];
#pragma unroll
  for (int r = 0; r < 8; ++r) {
    int q = q0 + r + half * 8;
    const int* mrow = mp + (size_t)q * S_ + kb + n16;
    float a0 = s0[r] * 0.125f;
    float a1 = s1[r] * 0.125f;
    if (__builtin_nontemporal_load(mrow) == 0) a0 = -1000.0f;
    if (__builtin_nontemporal_load(mrow + 16) == 0) a1 = -1000.0f;
    sv0[r] = a0;
    sv1[r] = a1;
  }

#pragma unroll
  for (int r = 0; r < 8; ++r) {
    float mx = fmaxf(sv0[r], sv1[r]);
#pragma unroll
    for (int off = 8; off; off >>= 1) mx = fmaxf(mx, __shfl_xor(mx, off, 32));
    float mnew = fmaxf(rowm[r], mx);
    float sc = __expf(rowm[r] - mnew);
    rowm[r] = mnew;
    float p0 = __expf(sv0[r] - mnew);
    float p1 = __expf(sv1[r] - mnew);
    float rs = p0 + p1;
#pragma unroll
    for (int off = 8; off; off >>= 1) rs += __shfl_xor(rs, off, 32);
    rowl[r] = rowl[r] * sc + rs;
#pragma unroll
    for (int t = 0; t < 4; ++t) o[t][r] *= sc;
    int mr = r + half * 8;
    pl[mr * 40 + n16] = (_Float16)p0;
    pl[mr * 40 + 16 + n16] = (_Float16)p1;
  }

  // reload P as A fragment (wave-private LDS region; DS in-order per wave)
  v16h ap = load_a16x32(pl, 40, 0, 0);
#pragma unroll
  for (int t = 0; t < 4; ++t) {
    v16h bv = load_b32x16(tvc, 40, t * 16, 0);
    o[t] = wmma32(ap, bv, o[t]);
  }
}

__global__ void k_attn(const _Float16* __restrict__ qh,
                       const _Float16* __restrict__ kh,
                       const _Float16* __restrict__ vt,
                       const int* __restrict__ mask,
                       _Float16* __restrict__ ctx) {
  __shared__ __align__(16) _Float16 ldsK[2 * KT_ELEM];
  __shared__ __align__(16) _Float16 ldsV[2 * VT_ELEM];
  __shared__ __align__(16) _Float16 ldsP[4 * 16 * 40];

  int tid = threadIdx.x;
  int lane = tid & 31;
  int wave = tid >> 5;
  int bh = blockIdx.y;
  int b = bh >> 4;
  int h = bh & (H_ - 1);
  int q0 = blockIdx.x * 64 + wave * 16;

  const _Float16* qp = qh + (size_t)bh * S_ * DH_;
  const _Float16* kp = kh + (size_t)bh * S_ * DH_;
  const _Float16* vp = vt + (size_t)bh * DH_ * S_;
  const int* mp = mask + (size_t)b * S_ * S_;
  _Float16* pl = ldsP + wave * (16 * 40);

  int half = lane >> 4, n16 = lane & 15;

  // per-thread async staging assignments (128 threads)
  int krow = tid >> 2, kcp = (tid & 3) * 2;       // K: 32 rows x 8 chunks
  const _Float16* kga = kp + (size_t)krow * DH_ + kcp * 8;
  unsigned int kl = (unsigned int)(uintptr_t)ldsK + (unsigned int)(krow * 160 + kcp * 16);
  int vrow = tid >> 1, vcp = (tid & 1) * 2;       // V: 64 rows x 4 chunks
  const _Float16* vga = vp + (size_t)vrow * S_ + vcp * 8;
  unsigned int vl = (unsigned int)(uintptr_t)ldsV + (unsigned int)(vrow * 80 + vcp * 16);

  const unsigned int KBYTES = KT_ELEM * 2;
  const unsigned int VBYTES = VT_ELEM * 2;

  v16h aql = load_a16x32(qp, DH_, q0, 0);
  v16h aqh2 = load_a16x32(qp, DH_, q0, 32);

  v8f o[4];
#pragma unroll
  for (int t = 0; t < 4; ++t) o[t] = zero8();
  float rowm[8], rowl[8];
#pragma unroll
  for (int r = 0; r < 8; ++r) { rowm[r] = -1e30f; rowl[r] = 0.0f; }

  // prologue: stage key-block 0 into buffer 0
  async_b128(kl, kga);
  async_b128(kl + 16, kga + 8);
  async_b128(vl, vga);
  async_b128(vl + 16, vga + 8);

  for (int it = 0; it < S_ / 32; it += 2) {
    // ---- half A: compute buf0 (key-block it), stage buf1 (it+1, always valid) ----
    wait_async0();
    __syncthreads();
    {
      const _Float16* kn = kga + (size_t)32 * DH_ * (it + 1);
      const _Float16* vn = vga + 32 * (it + 1);
      async_b128(kl + KBYTES, kn);
      async_b128(kl + KBYTES + 16, kn + 8);
      async_b128(vl + VBYTES, vn);
      async_b128(vl + VBYTES + 16, vn + 8);
    }
    attn_step(ldsK, ldsV, it * 32, q0, half, n16, mp, pl, aql, aqh2, o, rowm, rowl);

    // ---- half B: compute buf1 (key-block it+1), stage buf0 (it+2) ----
    wait_async0();
    __syncthreads();
    if (it + 2 < S_ / 32) {
      const _Float16* kn = kga + (size_t)32 * DH_ * (it + 2);
      const _Float16* vn = vga + 32 * (it + 2);
      async_b128(kl, kn);
      async_b128(kl + 16, kn + 8);
      async_b128(vl, vn);
      async_b128(vl + 16, vn + 8);
    }
    attn_step(ldsK + KT_ELEM, ldsV + VT_ELEM, (it + 1) * 32, q0, half, n16, mp, pl,
              aql, aqh2, o, rowm, rowl);
  }

#pragma unroll
  for (int r = 0; r < 8; ++r) {
    int q = q0 + r + half * 8;
    float inv = 1.0f / rowl[r];
    _Float16* cp = ctx + ((size_t)b * S_ + q) * D_ + h * DH_;
#pragma unroll
    for (int t = 0; t < 4; ++t) cp[t * 16 + n16] = (_Float16)(o[t][r] * inv);
  }
}

// ---------------- output projection GEMM ----------------

__global__ void k_gemm_out(const _Float16* __restrict__ ch,
                           const _Float16* __restrict__ wT,
                           const float* __restrict__ bias,
                           float* __restrict__ out) {
  __shared__ __align__(16) _Float16 tA[2 * TA_ELEM];
  __shared__ __align__(16) _Float16 tB[2 * TB_ELEM];

  int lane = threadIdx.x & 31;
  int wave = threadIdx.x >> 5;
  int mw = wave & 3, nw = wave >> 2;
  int m0blk = blockIdx.y * 64;
  int n0blk = blockIdx.x * 128;

  v8f acc[4];
#pragma unroll
  for (int t = 0; t < 4; ++t) acc[t] = zero8();

  gemm_mainloop(ch, wT, m0blk, n0blk, tA, tB, mw, nw, acc);

  int half = lane >> 4, n16 = lane & 15;
  int m0 = m0blk + mw * 16;
  int n0 = n0blk + nw * 64;
#pragma unroll
  for (int t = 0; t < 4; ++t) {
    int n = n0 + t * 16 + n16;
    float bv = bias[n];
#pragma unroll
    for (int r = 0; r < 8; ++r) {
      int m = m0 + r + half * 8;
      out[(size_t)m * D_ + n] = acc[t][r] + bv;
    }
  }
}

// ---------------- launch ----------------

extern "C" void kernel_launch(void* const* d_in, const int* in_sizes, int n_in,
                              void* d_out, int out_size, void* d_ws, size_t ws_size,
                              hipStream_t stream) {
  (void)in_sizes; (void)n_in; (void)out_size; (void)ws_size;

  const float* x = (const float*)d_in[0];
  const int* mask = (const int*)d_in[1];
  const float* w_qkv = (const float*)d_in[2];
  const float* b_qkv = (const float*)d_in[3];
  const float* w_out = (const float*)d_in[4];
  const float* b_out = (const float*)d_in[5];
  float* out = (float*)d_out;

  const size_t NX = (size_t)B_ * S_ * D_;          // 8388608
  const size_t NWQ = (size_t)D_ * 3 * D_;          // 3145728
  const size_t NWO = (size_t)D_ * D_;              // 1048576
  const size_t NQ = (size_t)B_ * H_ * S_ * DH_;    // 8388608

  _Float16* p = (_Float16*)d_ws;
  _Float16* x_h   = p; p += NX;
  _Float16* wqkvT = p; p += NWQ;
  _Float16* woutT = p; p += NWO;
  _Float16* q_h   = p; p += NQ;
  _Float16* k_h   = p; p += NQ;
  _Float16* v_t   = p; p += NQ;
  _Float16* ctx_h = p; p += NX;

  k_cvt_f16<<<(int)(NX / 256), 256, 0, stream>>>(x, x_h, (int)NX);
  k_transpose_f16<<<(int)(NWQ / 256), 256, 0, stream>>>(w_qkv, wqkvT, D_, 3 * D_);
  k_transpose_f16<<<(int)(NWO / 256), 256, 0, stream>>>(w_out, woutT, D_, D_);

  {
    dim3 grid(3 * D_ / 128, (B_ * S_) / 64);
    k_gemm_qkv<<<grid, 256, 0, stream>>>(x_h, wqkvT, b_qkv, q_h, k_h, v_t);
  }
  {
    dim3 grid(S_ / 64, B_ * H_);
    k_attn<<<grid, 128, 0, stream>>>(q_h, k_h, v_t, mask, ctx_h);
  }
  {
    dim3 grid(D_ / 128, (B_ * S_) / 64);
    k_gemm_out<<<grid, 256, 0, stream>>>(ctx_h, woutT, b_out, out);
  }
}
